// MixtureOfExperts_25872882992018
// MI455X (gfx1250) — matrix operand order repaired and checked by reading, not compiled, not run on hardware
//
#include <hip/hip_runtime.h>
#include <cstdint>
#include <cstddef>

// ---------------------------------------------------------------------------
// MoE top-2 forward for MI455X (gfx1250, wave32, WMMA bf16 16x16x32).
// ---------------------------------------------------------------------------

typedef __attribute__((ext_vector_type(16))) __bf16       v16bf;
typedef __attribute__((ext_vector_type(8)))  float        v8f;
typedef __attribute__((ext_vector_type(4)))  unsigned int v4u;

union Frag { v4u u[2]; v16bf v; };   // 32 bytes = 16 bf16 = one WMMA A/B operand

constexpr int T     = 8192;
constexpr int E     = 8;
constexpr int D_IN  = 1024;
constexpr int D_H   = 4096;
constexpr int D_OUT = 1024;
constexpr int CAP   = 8192;           // per-expert token capacity (upper bound)
constexpr int HCHUNK = 256;           // hidden-dim chunk held in LDS per iteration
constexpr int HSTR   = HCHUNK + 8;    // LDS row stride (bf16 units): 264*2B/4 = 132 words
                                      // -> bank(row m) = 4m mod 64: conflict-free b128 reads
constexpr int NHC    = D_H / HCHUNK;  // 16

// workspace layout (bytes)
constexpr size_t OFF_COUNTS = 0;                                   // 8 ints
constexpr size_t OFF_TOPI   = 1024;                                // T*2 ints
constexpr size_t OFF_TOPW   = OFF_TOPI + (size_t)T * 2 * 4;        // T*2 floats
constexpr size_t OFF_LISTS  = OFF_TOPW + (size_t)T * 2 * 4;        // E*CAP ints
constexpr size_t OFF_GWS    = OFF_LISTS + (size_t)E * CAP * 4;     // E*CAP floats
constexpr size_t OFF_XB     = (size_t)1 << 20;                     // T*D_IN bf16
constexpr size_t OFF_W1T    = OFF_XB  + (size_t)T * D_IN * 2;      // E*D_H*D_IN bf16 ([h][i])
constexpr size_t OFF_W2T    = OFF_W1T + (size_t)E * D_IN * D_H * 2;// E*D_OUT*D_H bf16 ([o][h])

__device__ __forceinline__ unsigned short f2bf(float f) {
  unsigned int u = __float_as_uint(f);
  u = (u + 0x7FFFu + ((u >> 16) & 1u)) >> 16;   // round-to-nearest-even
  return (unsigned short)u;
}

// --------------------------- fp32 -> bf16 copy -----------------------------
__global__ void k_convert_x(const float* __restrict__ x, unsigned short* __restrict__ xb) {
  size_t g = (size_t)blockIdx.x * 256 + threadIdx.x;
  xb[g] = f2bf(x[g]);
}

// ------------------- fp32 -> bf16 tiled transpose (per expert) -------------
// src: [R][C] fp32, dst: [C][R] bf16, grid = (C/32, R/32, E)
__global__ void k_transpose_bf16(const float* __restrict__ src, unsigned short* __restrict__ dst,
                                 int R, int C) {
  __shared__ float tile[32][33];
  const size_t eoff = (size_t)blockIdx.z * R * C;
  const float* s = src + eoff;
  unsigned short* d = dst + eoff;
  const int c0 = blockIdx.x * 32, r0 = blockIdx.y * 32;
  const int tx = threadIdx.x & 31, ty = threadIdx.x >> 5;   // 32 x 8
#pragma unroll
  for (int i = 0; i < 32; i += 8)
    tile[ty + i][tx] = s[(size_t)(r0 + ty + i) * C + (c0 + tx)];
  __syncthreads();
#pragma unroll
  for (int i = 0; i < 32; i += 8)
    d[(size_t)(c0 + ty + i) * R + (r0 + tx)] = f2bf(tile[tx][ty + i]);
}

// ------------------------------- gating ------------------------------------
// one wave per token; lanes split K; shfl reduce; lane 0 does top-2 + routing.
__global__ void k_gate(const float* __restrict__ x, const float* __restrict__ Wg,
                       const float* __restrict__ bg, int* __restrict__ counts,
                       int* __restrict__ topi, float* __restrict__ topw,
                       int* __restrict__ lists, float* __restrict__ gws) {
  const int wave = threadIdx.x >> 5, lane = threadIdx.x & 31;
  const int t = blockIdx.x * 8 + wave;
  const float* xr = x + (size_t)t * D_IN;
  float p[E] = {};
  for (int k = lane; k < D_IN; k += 32) {
    const float xv = xr[k];
    const float* wr = Wg + k * E;
#pragma unroll
    for (int e = 0; e < E; ++e) p[e] += xv * wr[e];
  }
#pragma unroll
  for (int off = 16; off; off >>= 1)
#pragma unroll
    for (int e = 0; e < E; ++e) p[e] += __shfl_xor(p[e], off, 32);

  if (lane == 0) {
    float v0 = -3.4e38f, v1 = -3.4e38f;
    int i0 = 0, i1 = 0;
#pragma unroll
    for (int e = 0; e < E; ++e) {
      const float lv = p[e] + bg[e];
      if (lv > v0)      { v1 = v0; i1 = i0; v0 = lv; i0 = e; }
      else if (lv > v1) { v1 = lv; i1 = e; }
    }
    const float ex = __expf(v1 - v0);
    const float s  = 1.0f + ex;
    const float w0 = 1.0f / s, w1 = ex / s;
    topi[2 * t] = i0; topi[2 * t + 1] = i1;
    topw[2 * t] = w0; topw[2 * t + 1] = w1;
    int p0 = atomicAdd(&counts[i0], 1);
    lists[i0 * CAP + p0] = t;  gws[i0 * CAP + p0] = w0;
    int p1 = atomicAdd(&counts[i1], 1);
    lists[i1 * CAP + p1] = t;  gws[i1 * CAP + p1] = w1;
  }
}

// ----------------------- output init with b2 combo -------------------------
__global__ void k_bias_init(const int* __restrict__ topi, const float* __restrict__ topw,
                            const float* __restrict__ b2, float* __restrict__ out) {
  const size_t g = (size_t)blockIdx.x * 256 + threadIdx.x;
  const int t = (int)(g >> 10), o = (int)(g & 1023);
  out[g] = topw[2 * t]     * b2[(size_t)topi[2 * t]     * D_OUT + o] +
           topw[2 * t + 1] * b2[(size_t)topi[2 * t + 1] * D_OUT + o];
}

// --------------------------- grouped expert GEMM ---------------------------
// block = 256 threads (8 waves) handles 32 tokens of expert blockIdx.x.
__global__ __launch_bounds__(256, 1)
void k_expert(const unsigned short* __restrict__ Xb,
              const unsigned short* __restrict__ W1T,   // [E][D_H][D_IN] bf16
              const unsigned short* __restrict__ W2T,   // [E][D_OUT][D_H] bf16
              const float* __restrict__ b1,             // [E][D_H]
              const int* __restrict__ counts,
              const int* __restrict__ lists,
              const float* __restrict__ gws,
              float* __restrict__ out) {
  const int e = blockIdx.x;
  const int n_e = counts[e];
  const int base = blockIdx.y * 32;
  if (base >= n_e) return;

  __shared__ unsigned short Hc[32 * HSTR];   // 16.9 KB bf16 hidden tile
  __shared__ int   s_tok[32];
  __shared__ float s_w[32];

  const int tid = threadIdx.x;
  if (tid < 32) {
    const int p = base + tid;
    s_tok[tid] = (p < n_e) ? lists[e * CAP + p] : 0;
    s_w[tid]   = (p < n_e) ? gws[e * CAP + p]   : 0.0f;
  }
  __syncthreads();

  const int wave = tid >> 5, lane = tid & 31;
  const int mt = wave & 1;          // which 16-row M tile (of 32 tokens)
  const int wq = wave >> 1;         // 0..3 column-quadrant owner
  const int lm = lane & 15;
  const int lh = lane >> 4;

  const unsigned short* W1e = W1T + (size_t)e * D_H * D_IN;
  const unsigned short* W2e = W2T + (size_t)e * D_OUT * D_H;
  const float* b1e = b1 + (size_t)e * D_H;
  const unsigned short* xrow = Xb + (size_t)s_tok[mt * 16 + lm] * D_IN;

  const v8f vzero = {0.f, 0.f, 0.f, 0.f, 0.f, 0.f, 0.f, 0.f};
  v8f acc[16];                      // 16 output tiles (wq*16+j), persistent f32
#pragma unroll
  for (int j = 0; j < 16; ++j) acc[j] = vzero;

#pragma unroll 1
  for (int hc = 0; hc < NHC; ++hc) {
    const int hbase = hc * HCHUNK;

    // ---- phase 1: Hc[32][HCHUNK] = relu(X32 @ W1[:, hbase..] + b1) ----
    v8f c1[4];
#pragma unroll
    for (int j2 = 0; j2 < 4; ++j2) c1[j2] = vzero;

#pragma unroll 1
    for (int kt = 0; kt < D_IN / 32; ++kt) {       // 32 K-steps of 32
      Frag a;
      const int kb = kt * 32 + lh * 8;
      a.u[0] = *(const v4u*)(xrow + kb);
      a.u[1] = *(const v4u*)(xrow + kb + 16);
      const int kbb = kt * 32 + lh * 8;
#pragma unroll
      for (int j2 = 0; j2 < 4; ++j2) {
        const int hcol = hbase + (wq * 4 + j2) * 16 + lm;   // B column (hidden)
        Frag b;
        b.u[0] = *(const v4u*)(W1e + (size_t)hcol * D_IN + kbb);
        b.u[1] = *(const v4u*)(W1e + (size_t)hcol * D_IN + kbb + 16);
        c1[j2] = __builtin_amdgcn_wmma_f32_16x16x32_bf16(
            false, a.v, false, b.v, (short)0, c1[j2], false, false);
      }
    }
#pragma unroll
    for (int j2 = 0; j2 < 4; ++j2) {
      const int ntl = wq * 4 + j2;
      const float bv = b1e[hbase + ntl * 16 + lm];
#pragma unroll
      for (int v = 0; v < 8; ++v) {
        float h = c1[j2][v] + bv;
        h = h > 0.f ? h : 0.f;
        const int mrow = mt * 16 + v + 8 * lh;    // D row mapping
        Hc[mrow * HSTR + ntl * 16 + lm] = f2bf(h);
      }
    }
    __syncthreads();

    // ---- phase 2: acc += Hc @ W2[hbase.., :] ----
    const unsigned short* hrow = &Hc[(mt * 16 + lm) * HSTR];
#pragma unroll 1
    for (int kt = 0; kt < HCHUNK / 32; ++kt) {     // 8 K-steps of 32
      Frag a;
      const int kb = kt * 32 + lh * 8;
      a.u[0] = *(const v4u*)(hrow + kb);           // ds_load_b128, bank-skewed
      a.u[1] = *(const v4u*)(hrow + kb + 16);
      const int kbb = hbase + kt * 32 + lh * 8;
#pragma unroll
      for (int j = 0; j < 16; ++j) {
        const int ocol = (wq * 16 + j) * 16 + lm;  // B column (output)
        Frag b;
        b.u[0] = *(const v4u*)(W2e + (size_t)ocol * D_H + kbb);
        b.u[1] = *(const v4u*)(W2e + (size_t)ocol * D_H + kbb + 16);
        acc[j] = __builtin_amdgcn_wmma_f32_16x16x32_bf16(
            false, a.v, false, b.v, (short)0, acc[j], false, false);
      }
    }
    __syncthreads();
  }

  // ---- gate-weighted scatter-add into out ----
  float wv[8]; int tv[8];
#pragma unroll
  for (int v = 0; v < 8; ++v) {
    const int mrow = mt * 16 + v + 8 * lh;
    wv[v] = s_w[mrow];
    tv[v] = s_tok[mrow];
  }
#pragma unroll
  for (int j = 0; j < 16; ++j) {
    const int ocol = (wq * 16 + j) * 16 + lm;
#pragma unroll
    for (int v = 0; v < 8; ++v) {
      if (wv[v] != 0.0f)
        atomicAdd(out + (size_t)tv[v] * D_OUT + ocol, wv[v] * acc[j][v]);
    }
  }
}

// ---------------------------------------------------------------------------
extern "C" void kernel_launch(void* const* d_in, const int* in_sizes, int n_in,
                              void* d_out, int out_size, void* d_ws, size_t ws_size,
                              hipStream_t stream) {
  (void)in_sizes; (void)n_in; (void)out_size; (void)ws_size;
  const float* x  = (const float*)d_in[0];
  const float* Wg = (const float*)d_in[1];
  const float* bg = (const float*)d_in[2];
  const float* W1 = (const float*)d_in[3];
  const float* b1 = (const float*)d_in[4];
  const float* W2 = (const float*)d_in[5];
  const float* b2 = (const float*)d_in[6];
  float* out = (float*)d_out;

  char* ws = (char*)d_ws;
  int*            counts = (int*)(ws + OFF_COUNTS);
  int*            topi   = (int*)(ws + OFF_TOPI);
  float*          topw   = (float*)(ws + OFF_TOPW);
  int*            lists  = (int*)(ws + OFF_LISTS);
  float*          gws    = (float*)(ws + OFF_GWS);
  unsigned short* xb     = (unsigned short*)(ws + OFF_XB);
  unsigned short* w1t    = (unsigned short*)(ws + OFF_W1T);
  unsigned short* w2t    = (unsigned short*)(ws + OFF_W2T);

  hipMemsetAsync(counts, 0, E * sizeof(int), stream);

  k_convert_x<<<(T * D_IN) / 256, 256, 0, stream>>>(x, xb);
  k_transpose_bf16<<<dim3(D_H / 32, D_IN / 32, E), 256, 0, stream>>>(W1, w1t, D_IN, D_H);
  k_transpose_bf16<<<dim3(D_OUT / 32, D_H / 32, E), 256, 0, stream>>>(W2, w2t, D_H, D_OUT);
  k_gate<<<T / 8, 256, 0, stream>>>(x, Wg, bg, counts, topi, topw, lists, gws);
  k_bias_init<<<(T * D_OUT) / 256, 256, 0, stream>>>(topi, topw, b2, out);
  k_expert<<<dim3(E, CAP / 32), 256, 0, stream>>>(xb, w1t, w2t, b1, counts, lists, gws, out);
}
